// CovNerf_26130581029079
// MI455X (gfx1250) — matrix operand order repaired and checked by reading, not compiled, not run on hardware
//
#include <hip/hip_runtime.h>
#include <math.h>

// ---------------- CDNA5 wave32 / WMMA types ----------------
typedef __attribute__((ext_vector_type(16))) __bf16 v16bf;
typedef __attribute__((ext_vector_type(8)))  __bf16 v8bf;
typedef __attribute__((ext_vector_type(8)))  float  v8f;

#define SAMP 64          // n_ray_samples (reference default; scalar input not host-readable)
#define NBANDS 6

static __device__ __forceinline__ int imin(int a, int b) { return a < b ? a : b; }

static __device__ __forceinline__ v16bf cat16(v8bf a, v8bf b) {
    return __builtin_shufflevector(a, b, 0,1,2,3,4,5,6,7,8,9,10,11,12,13,14,15);
}

// =====================================================================
// Weight packing: fp32 (K x N, optionally 3 concatenated along N) ->
// bf16 in the exact V_WMMA_F32_16X16X32_BF16 B-fragment layout:
//   frag(kt,nt): 32 lanes x 16 bf16 contiguous;
//   lane L: col = nt*16 + (L&15), k = kt*32 + (L<16?0:8) + (j&7) + (j>=8?16:0)   (j=0..15)
// =====================================================================
__global__ void pack_w(const float* __restrict__ W0, const float* __restrict__ W1,
                       const float* __restrict__ W2,
                       int K, int N0, int N1, int N2, int Kp, int Npad,
                       __bf16* __restrict__ out)
{
    int idx = blockIdx.x * blockDim.x + threadIdx.x;
    int total = (Kp >> 5) * (Npad >> 4) * 512;
    if (idx >= total) return;
    int j    = idx & 15;
    int lane = (idx >> 4) & 31;
    int tile = idx >> 9;
    int ntiles = Npad >> 4;
    int nt = tile % ntiles;
    int kt = tile / ntiles;
    int n = nt * 16 + (lane & 15);
    int k = kt * 32 + ((lane < 16) ? 0 : 8) + (j & 7) + ((j >> 3) << 4);
    float val = 0.f;
    int Ntot = N0 + N1 + N2;
    if (k < K && n < Ntot) {
        if (n < N0)            val = W0[(size_t)k * N0 + n];
        else if (n < N0 + N1)  val = W1[(size_t)k * N1 + (n - N0)];
        else                   val = W2[(size_t)k * N2 + (n - N0 - N1)];
    }
    out[idx] = (__bf16)val;
}

// =====================================================================
// Featurizer: per (ray-local rl, sample s, view v) token computes the
// 83-channel input (bilinear rgb, bilinear feats, posenc(viewdir),
// posenc(t)), padded to 96, stored row-major bf16 (WMMA A layout reads
// two contiguous 16B chunks per lane from this).
// =====================================================================
__global__ void featurize(const float* __restrict__ rgbs, const float* __restrict__ feats,
                          const float* __restrict__ rays_o, const float* __restrict__ rays_d,
                          const float* __restrict__ intr, const float* __restrict__ cfr,
                          int r0, int rc, int V, int Hh, int Ww, int Hf, int Wf,
                          __bf16* __restrict__ X0, float* __restrict__ tb)
{
    int tid = blockIdx.x * blockDim.x + threadIdx.x;
    int total = rc * SAMP * V;
    if (tid >= total) return;
    int v  = tid % V;
    int s  = (tid / V) % SAMP;
    int rl = tid / (V * SAMP);
    int r  = r0 + rl;

    float ox = rays_o[r * 3 + 0], oy = rays_o[r * 3 + 1], oz = rays_o[r * 3 + 2];
    float dx = rays_d[r * 3 + 0], dy = rays_d[r * 3 + 1], dz = rays_d[r * 3 + 2];
    float tn = (2.0f - oz) / dz, tf = (6.0f - oz) / dz;
    float u  = (s + 0.5f) / (float)SAMP;
    float t  = tn + (tf - tn) * u;
    float px = ox + t * dx, py = oy + t * dy, pz = oz + t * dz;

    const float* M4 = cfr + v * 16;                 // 4x4 row-major cam_from_ref
    float cx = M4[0] * px + M4[1] * py + M4[2]  * pz + M4[3];
    float cy = M4[4] * px + M4[5] * py + M4[6]  * pz + M4[7];
    float cz = M4[8] * px + M4[9] * py + M4[10] * pz + M4[11];
    const float* Km = intr + v * 9;
    float uu = Km[0] * cx + Km[1] * cy + Km[2] * cz;
    float vv = Km[3] * cx + Km[4] * cy + Km[5] * cz;
    float ww = Km[6] * cx + Km[7] * cy + Km[8] * cz;
    float wz = fmaxf(ww, 0.001f);
    float pxx = uu / wz, pyy = vv / wz;

    __bf16* row = X0 + (size_t)((rl * SAMP + s) * V + v) * 96;

    { // rgb (3 channels, HxW)
        float xq = fminf(fmaxf(pxx, 0.f), (float)(Ww - 1));
        float yq = fminf(fmaxf(pyy, 0.f), (float)(Hh - 1));
        float x0f = floorf(xq), y0f = floorf(yq);
        int x0 = (int)x0f, y0 = (int)y0f;
        int x1 = imin(x0 + 1, Ww - 1), y1 = imin(y0 + 1, Hh - 1);
        float wx = xq - x0f, wy = yq - y0f;
        const float* img = rgbs + (size_t)v * 3 * Hh * Ww;
        #pragma unroll
        for (int c = 0; c < 3; ++c) {
            const float* pl = img + (size_t)c * Hh * Ww;
            float val = pl[y0 * Ww + x0] * (1 - wy) * (1 - wx) + pl[y0 * Ww + x1] * (1 - wy) * wx
                      + pl[y1 * Ww + x0] * wy * (1 - wx)       + pl[y1 * Ww + x1] * wy * wx;
            row[c] = (__bf16)val;
        }
    }
    { // feats (32 channels, Hf x Wf), coords/4, values/4
        float xs = pxx * 0.25f, ys = pyy * 0.25f;
        float xq = fminf(fmaxf(xs, 0.f), (float)(Wf - 1));
        float yq = fminf(fmaxf(ys, 0.f), (float)(Hf - 1));
        float x0f = floorf(xq), y0f = floorf(yq);
        int x0 = (int)x0f, y0 = (int)y0f;
        int x1 = imin(x0 + 1, Wf - 1), y1 = imin(y0 + 1, Hf - 1);
        float wx = xq - x0f, wy = yq - y0f;
        float w00 = (1 - wy) * (1 - wx) * 0.25f, w01 = (1 - wy) * wx * 0.25f;
        float w10 = wy * (1 - wx) * 0.25f,       w11 = wy * wx * 0.25f;
        const float* img = feats + (size_t)v * 32 * Hf * Wf;
        #pragma unroll
        for (int c = 0; c < 32; ++c) {
            const float* pl = img + (size_t)c * Hf * Wf;
            float val = pl[y0 * Wf + x0] * w00 + pl[y0 * Wf + x1] * w01
                      + pl[y1 * Wf + x0] * w10 + pl[y1 * Wf + x1] * w11;
            row[3 + c] = (__bf16)val;
        }
    }
    { // posenc(viewdir): 3 comps * (6 sin, 6 cos) = 36 at offset 35
        float wxv = px - M4[3], wyv = py - M4[7], wzv = pz - M4[11];
        float inv = rsqrtf(wxv * wxv + wyv * wyv + wzv * wzv);
        float vd[3] = { wxv * inv, wyv * inv, wzv * inv };
        #pragma unroll
        for (int i = 0; i < 3; ++i) {
            float f = 1.f;
            #pragma unroll
            for (int b = 0; b < NBANDS; ++b) {
                float a = vd[i] * f;
                row[35 + i * 12 + b]     = (__bf16)sinf(a);
                row[35 + i * 12 + 6 + b] = (__bf16)cosf(a);
                f *= 2.f;
            }
        }
    }
    { // posenc(t): 6 sin + 6 cos at offset 71
        float f = 1.f;
        #pragma unroll
        for (int b = 0; b < NBANDS; ++b) {
            float a = t * f;
            row[71 + b] = (__bf16)sinf(a);
            row[77 + b] = (__bf16)cosf(a);
            f *= 2.f;
        }
    }
    #pragma unroll
    for (int c = 83; c < 96; ++c) row[c] = (__bf16)0.f;   // K pad
    if (v == 0) tb[rl * SAMP + s] = t;
}

// =====================================================================
// WMMA GEMM: out[M x N] = A[M x Kp](bf16,row-major) * Bpacked + bias
//            (+ residual fp32) (+ leaky-relu), write fp32 and/or bf16.
// Block = 128 threads = 4 waves; block tile 128(M) x NSUB*16(N).
// Each wave owns 32(M) x NSUB*16(N): 2 A fragments + NSUB B fragments,
// 2*NSUB WMMAs per K-step, software-pipelined (next K-step fragments are
// fetched before this step's WMMAs issue). NSUB is a template constant so
// the inner loop is branch-free straight-line code.
// M must be a multiple of 32 (launcher passes multiples of SAMP=64).
// =====================================================================
template <int NSUB>
__global__ __launch_bounds__(128)
void gemm_wmma(const __bf16* __restrict__ A, const __bf16* __restrict__ Bp,
               const float* __restrict__ bias, const float* __restrict__ resid,
               float* __restrict__ outF, __bf16* __restrict__ outH,
               int M, int Kp, int N, int act)
{
    int lane = threadIdx.x & 31;
    int wv   = threadIdx.x >> 5;
    int m0   = blockIdx.y * 128 + wv * 32;
    int n0   = blockIdx.x * (NSUB * 16);
    if (m0 >= M) return;
    int ktiles  = Kp >> 5;
    int ntilesN = N >> 4;

    v8f acc[2][NSUB] = {};
    int c0 = (lane < 16) ? 0 : 8;         // 16-bit A-fragment chunk start
    const __bf16* ar0 = A + (size_t)(m0 + (lane & 15)) * Kp;
    const __bf16* ar1 = ar0 + (size_t)16 * Kp;
    const __bf16* bbase = Bp + ((size_t)(n0 >> 4) * 32 + lane) * 16;

    auto loadA = [&](int kt, v16bf& x0, v16bf& x1) {
        x0 = cat16(*(const v8bf*)(ar0 + kt * 32 + c0), *(const v8bf*)(ar0 + kt * 32 + c0 + 16));
        x1 = cat16(*(const v8bf*)(ar1 + kt * 32 + c0), *(const v8bf*)(ar1 + kt * 32 + c0 + 16));
    };
    auto loadB = [&](int kt, v16bf* b) {
        #pragma unroll
        for (int j = 0; j < NSUB; ++j)
            b[j] = *(const v16bf*)(bbase + ((size_t)kt * ntilesN + j) * 512);
    };

    v16bf a0, a1, b[NSUB];
    loadA(0, a0, a1);
    loadB(0, b);
    for (int kt = 0; kt < ktiles; ++kt) {
        v16bf a0n = a0, a1n = a1, bn[NSUB];
        #pragma unroll
        for (int j = 0; j < NSUB; ++j) bn[j] = b[j];
        if (kt + 1 < ktiles) {            // prefetch next K-step into regs
            loadA(kt + 1, a0n, a1n);
            loadB(kt + 1, bn);
        }
        #pragma unroll
        for (int j = 0; j < NSUB; ++j) {
            acc[0][j] = __builtin_amdgcn_wmma_f32_16x16x32_bf16(
                            false, a0, false, b[j], (short)0, acc[0][j], false, false);
            acc[1][j] = __builtin_amdgcn_wmma_f32_16x16x32_bf16(
                            false, a1, false, b[j], (short)0, acc[1][j], false, false);
        }
        a0 = a0n; a1 = a1n;
        #pragma unroll
        for (int j = 0; j < NSUB; ++j) b[j] = bn[j];
    }

    // Epilogue: C/D layout -> VGPR r holds (M = m0+f*16+r+(lane<16?0:8), N = n0+j*16+(lane&15))
    #pragma unroll
    for (int f = 0; f < 2; ++f) {
        #pragma unroll
        for (int j = 0; j < NSUB; ++j) {
            int col = n0 + j * 16 + (lane & 15);
            float bval = bias ? bias[col] : 0.f;
            #pragma unroll
            for (int r = 0; r < 8; ++r) {
                int orow = m0 + f * 16 + r + ((lane < 16) ? 0 : 8);
                size_t idx = (size_t)orow * N + col;
                float val = acc[f][j][r] + bval;
                if (resid) val += resid[idx];
                if (act == 1) val = (val > 0.f) ? val : 0.01f * val;    // leaky relu
                if (outF) outF[idx] = val;
                if (outH) outH[idx] = (__bf16)val;
            }
        }
    }
}

// ---------------- LayerNorm over 256 channels: one wave per token ----------------
__global__ __launch_bounds__(256)
void ln_k(const float* __restrict__ x, __bf16* __restrict__ o, int M)
{
    int gt = blockIdx.x * blockDim.x + threadIdx.x;
    int tok = gt >> 5, lane = gt & 31;
    if (tok >= M) return;
    const float* row = x + (size_t)tok * 256;
    float v[8]; float s = 0.f;
    #pragma unroll
    for (int i = 0; i < 8; ++i) { v[i] = row[i * 32 + lane]; s += v[i]; }
    #pragma unroll
    for (int off = 16; off >= 1; off >>= 1) s += __shfl_xor(s, off, 32);
    float mean = s * (1.f / 256.f);
    float q = 0.f;
    #pragma unroll
    for (int i = 0; i < 8; ++i) { float d = v[i] - mean; q += d * d; }
    #pragma unroll
    for (int off = 16; off >= 1; off >>= 1) q += __shfl_xor(q, off, 32);
    float inv = rsqrtf(q * (1.f / 256.f) + 1e-5f);
    __bf16* orow = o + (size_t)tok * 256;
    #pragma unroll
    for (int i = 0; i < 8; ++i) orow[i * 32 + lane] = (__bf16)((v[i] - mean) * inv);
}

// ---------------- Attention over V=3 views (per (r,s) group, per head) ----------------
__global__ void attn_v(const __bf16* __restrict__ qkv, __bf16* __restrict__ o, int G)
{
    int t = blockIdx.x * blockDim.x + threadIdx.x;
    if (t >= G * 8) return;
    int h = t & 7, g = t >> 3;
    float q[3][16], k[3][16], w[3][16];
    #pragma unroll
    for (int m = 0; m < 3; ++m) {
        size_t base = ((size_t)(g * 3 + m)) * 384 + h * 16;
        #pragma unroll
        for (int d = 0; d < 16; ++d) {
            q[m][d] = (float)qkv[base + d];
            k[m][d] = (float)qkv[base + 128 + d];
            w[m][d] = (float)qkv[base + 256 + d];
        }
    }
    #pragma unroll
    for (int vq = 0; vq < 3; ++vq) {
        float sc[3]; float mx = -1e30f;
        #pragma unroll
        for (int m = 0; m < 3; ++m) {
            float s2 = 0.f;
            #pragma unroll
            for (int d = 0; d < 16; ++d) s2 += q[vq][d] * k[m][d];
            sc[m] = s2 * 0.25f;                 // 1/sqrt(16)
            mx = fmaxf(mx, sc[m]);
        }
        float sum = 0.f;
        #pragma unroll
        for (int m = 0; m < 3; ++m) { sc[m] = expf(sc[m] - mx); sum += sc[m]; }
        float is = 1.f / sum;
        __bf16* orow = o + ((size_t)(g * 3 + vq)) * 128 + h * 16;
        #pragma unroll
        for (int d = 0; d < 16; ++d) {
            float a2 = 0.f;
            #pragma unroll
            for (int m = 0; m < 3; ++m) a2 += sc[m] * w[m][d];
            orow[d] = (__bf16)(a2 * is);
        }
    }
}

// ---------------- Mean over V ----------------
__global__ void mean_v(const float* __restrict__ x, float* __restrict__ y, int G, int V)
{
    int idx = blockIdx.x * blockDim.x + threadIdx.x;
    if (idx >= G * 256) return;
    int g = idx >> 8, c = idx & 255;
    float s = 0.f;
    for (int vv = 0; vv < V; ++vv) s += x[((size_t)(g * V + vv)) * 256 + c];
    y[idx] = s / (float)V;
}

// ---------------- Attention over S=64 samples (per ray, per head), K/V staged in LDS ----------------
__global__ __launch_bounds__(SAMP)
void attn_ray(const __bf16* __restrict__ qkv, __bf16* __restrict__ o)
{
    __shared__ float kk[SAMP][16];
    __shared__ float vvs[SAMP][16];
    int h  = blockIdx.x & 7;
    int rl = blockIdx.x >> 3;
    int s  = threadIdx.x;
    size_t rowq = ((size_t)(rl * SAMP + s)) * 384 + h * 16;
    float q[16];
    #pragma unroll
    for (int d = 0; d < 16; ++d) {
        kk[s][d]  = (float)qkv[rowq + 128 + d];
        vvs[s][d] = (float)qkv[rowq + 256 + d];
        q[d]      = (float)qkv[rowq + d];
    }
    __syncthreads();
    float mx = -1e30f;
    for (int m = 0; m < SAMP; ++m) {
        float sc = 0.f;
        #pragma unroll
        for (int d = 0; d < 16; ++d) sc += q[d] * kk[m][d];
        mx = fmaxf(mx, sc * 0.25f);
    }
    float sum = 0.f, acc[16] = {};
    for (int m = 0; m < SAMP; ++m) {
        float sc = 0.f;
        #pragma unroll
        for (int d = 0; d < 16; ++d) sc += q[d] * kk[m][d];
        float wgt = expf(sc * 0.25f - mx);
        sum += wgt;
        #pragma unroll
        for (int d = 0; d < 16; ++d) acc[d] += wgt * vvs[m][d];
    }
    float is = 1.f / sum;
    __bf16* orow = o + ((size_t)(rl * SAMP + s)) * 128 + h * 16;
    #pragma unroll
    for (int d = 0; d < 16; ++d) orow[d] = (__bf16)(acc[d] * is);
}

// ---------------- fp32 -> bf16 elementwise ----------------
__global__ void cvt_bf(const float* __restrict__ x, __bf16* __restrict__ o, int n)
{
    int i = blockIdx.x * blockDim.x + threadIdx.x;
    if (i < n) o[i] = (__bf16)x[i];
}

// ---------------- Final: per-ray softmax(alpha), sigmoid(rgb), write (t,a,r,g,b) ----------------
__global__ __launch_bounds__(SAMP)
void final_k(const float* __restrict__ logits, const float* __restrict__ b_out,
             const float* __restrict__ tb, float* __restrict__ out, int r0)
{
    __shared__ float red[SAMP];
    int s = threadIdx.x, rl = blockIdx.x;
    int n2 = rl * SAMP + s;
    float l0 = logits[(size_t)n2 * 16] + b_out[0];
    red[s] = l0; __syncthreads();
    for (int off = 32; off >= 1; off >>= 1) { if (s < off) red[s] = fmaxf(red[s], red[s + off]); __syncthreads(); }
    float mx = red[0]; __syncthreads();
    float e = expf(l0 - mx);
    red[s] = e; __syncthreads();
    for (int off = 32; off >= 1; off >>= 1) { if (s < off) red[s] += red[s + off]; __syncthreads(); }
    float alpha = e / red[0];
    size_t base = ((size_t)(r0 + rl) * SAMP + s) * 5;
    out[base + 0] = tb[n2];
    out[base + 1] = alpha;
    #pragma unroll
    for (int c = 1; c < 4; ++c) {
        float l = logits[(size_t)n2 * 16 + c] + b_out[c];
        out[base + 1 + c] = 1.f / (1.f + expf(-l));
    }
}

// =====================================================================
// Host launcher
// =====================================================================
extern "C" void kernel_launch(void* const* d_in, const int* in_sizes, int n_in,
                              void* d_out, int out_size, void* d_ws, size_t ws_size,
                              hipStream_t stream)
{
    const float* rgbs   = (const float*)d_in[0];
    const float* feats  = (const float*)d_in[1];
    const float* rays_o = (const float*)d_in[2];
    const float* rays_d = (const float*)d_in[3];
    const float* intr   = (const float*)d_in[4];
    const float* cfr    = (const float*)d_in[5];
    const float* W_in   = (const float*)d_in[6];
    const float* b_in   = (const float*)d_in[7];
    const float* nWq = (const float*)d_in[8],  *nWk = (const float*)d_in[9];
    const float* nWv = (const float*)d_in[10], *nWo = (const float*)d_in[11];
    const float* nW1 = (const float*)d_in[12], *nb1 = (const float*)d_in[13];
    const float* nW2 = (const float*)d_in[14], *nb2 = (const float*)d_in[15];
    const float* rWq = (const float*)d_in[16], *rWk = (const float*)d_in[17];
    const float* rWv = (const float*)d_in[18], *rWo = (const float*)d_in[19];
    const float* rW1 = (const float*)d_in[20], *rb1 = (const float*)d_in[21];
    const float* rW2 = (const float*)d_in[22], *rb2 = (const float*)d_in[23];
    const float* W_out = (const float*)d_in[24];
    const float* b_out = (const float*)d_in[25];
    (void)n_in; (void)out_size;

    int V = in_sizes[5] / 16;                  // cam_from_ref: V*4*4
    int R = in_sizes[2] / 3;                   // rays_o: R*3
    int HW = in_sizes[0] / (3 * V);
    int Hh = 256; for (int h = 16; h <= 4096; ++h) if (h * h == HW) { Hh = h; break; }
    int Ww = Hh, Hf = Hh / 4, Wf = Ww / 4;

    int Rc = 256; if (R < Rc) Rc = R;          // ray chunk (keeps workspace ~180MB)
    int M1max = Rc * SAMP * V;
    int M2max = Rc * SAMP;

    // ---- workspace carve (bump allocator, 256B aligned) ----
    char* wp = (char*)d_ws;
    auto alloc = [&](size_t bytes) { void* r = (void*)wp; wp += (bytes + 255) & ~(size_t)255; return r; };
    __bf16* wp_in   = (__bf16*)alloc((size_t)3 * 16 * 512 * 2);   // 96x256
    __bf16* wp_nqkv = (__bf16*)alloc((size_t)8 * 24 * 512 * 2);   // 256x384
    __bf16* wp_no   = (__bf16*)alloc((size_t)4 * 16 * 512 * 2);   // 128x256
    __bf16* wp_nw1  = (__bf16*)alloc((size_t)8 * 16 * 512 * 2);
    __bf16* wp_nw2  = (__bf16*)alloc((size_t)8 * 16 * 512 * 2);
    __bf16* wp_rqkv = (__bf16*)alloc((size_t)8 * 24 * 512 * 2);
    __bf16* wp_ro   = (__bf16*)alloc((size_t)4 * 16 * 512 * 2);
    __bf16* wp_rw1  = (__bf16*)alloc((size_t)8 * 16 * 512 * 2);
    __bf16* wp_rw2  = (__bf16*)alloc((size_t)8 * 16 * 512 * 2);
    __bf16* wp_out  = (__bf16*)alloc((size_t)8 * 1 * 512 * 2);    // 256x16 (N padded 4->16)
    __bf16* X0    = (__bf16*)alloc((size_t)M1max * 96 * 2);
    float*  tb    = (float*) alloc((size_t)M2max * 4);
    float*  x1    = (float*) alloc((size_t)M1max * 256 * 4);
    __bf16* lnb   = (__bf16*)alloc((size_t)M1max * 256 * 2);
    __bf16* qkvb  = (__bf16*)alloc((size_t)M1max * 384 * 2);      // also reused as MLP hidden (bf16)
    __bf16* ob    = (__bf16*)alloc((size_t)M1max * 128 * 2);
    float*  yb    = (float*) alloc((size_t)M2max * 256 * 4);
    __bf16* ln2   = (__bf16*)alloc((size_t)M2max * 256 * 2);
    __bf16* qkv2  = (__bf16*)alloc((size_t)M2max * 384 * 2);      // also reused as MLP hidden
    __bf16* o2    = (__bf16*)alloc((size_t)M2max * 128 * 2);
    float*  logit = (float*) alloc((size_t)M2max * 16 * 4);
    (void)ws_size;

    auto pack = [&](const float* a, const float* b2, const float* c2,
                    int K, int N0i, int N1i, int N2i, int Kp, int Npad, __bf16* out) {
        int total = (Kp >> 5) * (Npad >> 4) * 512;
        pack_w<<<(total + 255) / 256, 256, 0, stream>>>(a, b2, c2, K, N0i, N1i, N2i, Kp, Npad, out);
    };
    auto gemm = [&](const __bf16* A, const __bf16* Bp, const float* bias, const float* resid,
                    float* oF, __bf16* oH, int M, int Kp, int N, int act) {
        if (N == 16) {
            dim3 g(1, (M + 127) / 128);
            gemm_wmma<1><<<g, 128, 0, stream>>>(A, Bp, bias, resid, oF, oH, M, Kp, N, act);
        } else {
            dim3 g(N / 64, (M + 127) / 128);
            gemm_wmma<4><<<g, 128, 0, stream>>>(A, Bp, bias, resid, oF, oH, M, Kp, N, act);
        }
    };
    auto lnK = [&](const float* x, __bf16* o, int M) {
        ln_k<<<(M + 7) / 8, 256, 0, stream>>>(x, o, M);
    };

    // ---- pack all weights to WMMA B-fragment layout (bf16) ----
    pack(W_in, nullptr, nullptr, 83, 256, 0, 0, 96, 256, wp_in);
    pack(nWq, nWk, nWv, 256, 128, 128, 128, 256, 384, wp_nqkv);
    pack(nWo, nullptr, nullptr, 128, 256, 0, 0, 128, 256, wp_no);
    pack(nW1, nullptr, nullptr, 256, 256, 0, 0, 256, 256, wp_nw1);
    pack(nW2, nullptr, nullptr, 256, 256, 0, 0, 256, 256, wp_nw2);
    pack(rWq, rWk, rWv, 256, 128, 128, 128, 256, 384, wp_rqkv);
    pack(rWo, nullptr, nullptr, 128, 256, 0, 0, 128, 256, wp_ro);
    pack(rW1, nullptr, nullptr, 256, 256, 0, 0, 256, 256, wp_rw1);
    pack(rW2, nullptr, nullptr, 256, 256, 0, 0, 256, 256, wp_rw2);
    pack(W_out, nullptr, nullptr, 256, 4, 0, 0, 256, 16, wp_out);

    // ---- pipeline, chunked over rays ----
    for (int r0 = 0; r0 < R; r0 += Rc) {
        int rc = (R - r0 < Rc) ? (R - r0) : Rc;
        int M1 = rc * SAMP * V;
        int M2 = rc * SAMP;

        featurize<<<(M1 + 255) / 256, 256, 0, stream>>>(rgbs, feats, rays_o, rays_d, intr, cfr,
                                                        r0, rc, V, Hh, Ww, Hf, Wf, X0, tb);
        // x = X @ W_in + b_in
        gemm(X0, wp_in, b_in, nullptr, x1, nullptr, M1, 96, 256, 0);
        // nerf block: attention over V
        lnK(x1, lnb, M1);
        gemm(lnb, wp_nqkv, nullptr, nullptr, nullptr, qkvb, M1, 256, 384, 0);
        attn_v<<<(M2 * 8 + 255) / 256, 256, 0, stream>>>(qkvb, ob, M2);
        gemm(ob, wp_no, nullptr, x1, x1, nullptr, M1, 128, 256, 0);        // x += o@Wo
        // nerf block: MLP
        lnK(x1, lnb, M1);
        gemm(lnb, wp_nw1, nb1, nullptr, nullptr, qkvb, M1, 256, 256, 1);   // leaky, bf16 hidden
        gemm(qkvb, wp_nw2, nb2, x1, x1, nullptr, M1, 256, 256, 0);         // x += h@W2 + b2
        // mean over V
        mean_v<<<(M2 * 256 + 255) / 256, 256, 0, stream>>>(x1, yb, M2, V);
        // ray block: attention over S
        lnK(yb, ln2, M2);
        gemm(ln2, wp_rqkv, nullptr, nullptr, nullptr, qkv2, M2, 256, 384, 0);
        attn_ray<<<rc * 8, SAMP, 0, stream>>>(qkv2, o2);
        gemm(o2, wp_ro, nullptr, yb, yb, nullptr, M2, 128, 256, 0);
        // ray block: MLP
        lnK(yb, ln2, M2);
        gemm(ln2, wp_rw1, rb1, nullptr, nullptr, qkv2, M2, 256, 256, 1);
        gemm(qkv2, wp_rw2, rb2, yb, yb, nullptr, M2, 256, 256, 0);
        // out head
        cvt_bf<<<(M2 * 256 + 255) / 256, 256, 0, stream>>>(yb, ln2, M2 * 256);
        gemm(ln2, wp_out, nullptr, nullptr, logit, nullptr, M2, 256, 16, 0);
        final_k<<<rc, SAMP, 0, stream>>>(logit, b_out, tb, (float*)d_out, r0);
    }
}
